// SNNQUT_91164975825526
// MI455X (gfx1250) — compile-verified, matrix operand order checked by reading, not run on hardware
//
#include <hip/hip_runtime.h>
#include <hip/hip_bf16.h>

// ---------------------------------------------------------------------------
// SNN forward scan on gfx1250 (MI455X).
//   - 1 wave (32 lanes) owns a 16-batch tile, runs the full T=2000 recurrence.
//   - GEMMs via v_wmma_f32_16x16x32_f16 (M=16 batch, N pad 24->32, K pad ->32).
//   - Weights + membrane state + spike state register-resident for all T.
//   - reset_t == spk_{t-1} (same comparator) -> no reset recompute per step.
//   - D->A layer transposes through a 1KB LDS tile (ds_store_b16 / ds_load_b128).
//   - x stream software-pipelined + global_prefetch ahead; output pointers
//     strength-reduced to per-step pointer bumps + immediate store offsets.
// ---------------------------------------------------------------------------

typedef __attribute__((ext_vector_type(16))) _Float16 v16h;
typedef __attribute__((ext_vector_type(8)))  _Float16 v8h;
typedef __attribute__((ext_vector_type(8)))  float    v8f;

#define SNN_B 512
#define SNN_T 2000
#define SNN_I 16
#define SNN_H 24
#define SNN_O 4

// Build a WMMA B-operand (32x16 f16, lane = N column, halves = K per ISA 16-bit
// layout): b[j] = W[n][8*hi + j], b[8+j] = W[n][16 + 8*hi + j], zero-padded.
__device__ inline v16h make_B(const float* __restrict__ W, int ldk, int kmax,
                              int nrows, int nbase) {
  const int lane = threadIdx.x & 31;
  const int n    = nbase + (lane & 15);
  const int k0   = (lane >> 4) * 8;
  v16h b;
#pragma unroll
  for (int j = 0; j < 8; ++j) {
    const int kA = k0 + j;
    const int kB = 16 + k0 + j;
    const float vA = (n < nrows && kA < kmax) ? W[(size_t)n * ldk + kA] : 0.0f;
    const float vB = (n < nrows && kB < kmax) ? W[(size_t)n * ldk + kB] : 0.0f;
    b[j]     = (_Float16)vA;
    b[8 + j] = (_Float16)vB;
  }
  return b;
}

// LIF with subtract-reset, thresh=1.  Key identity vs the reference:
//   reset_t = (m_{t-1} > 1)  ==  spk_{t-1} = (m_{t-1} - 1 > 0)
// so the previous step's spike register IS this step's reset.
//   m   = beta*m + cur - spk_prev
//   spk = (m > 1) ? 1 : 0
__device__ inline void lif_step(v8f& m, v8f& spk, const v8f cur, float beta) {
#pragma unroll
  for (int e = 0; e < 8; ++e) {
    const float mn = __builtin_fmaf(beta, m[e], cur[e]) - spk[e];
    m[e]   = mn;
    spk[e] = (mn > 1.0f) ? 1.0f : 0.0f;
  }
}

// Store one C/D-layout spike tile (lane = N, vgpr = M) into the LDS transpose
// buffer as f16 at [m][n], row stride 32 halves.
__device__ inline void spill_spikes(_Float16* sbuf, const v8f s, int ncolbase) {
  const int lane = threadIdx.x & 31;
  const int hi   = lane >> 4;
  const int ncol = ncolbase + (lane & 15);
#pragma unroll
  for (int r = 0; r < 8; ++r)
    sbuf[(r + 8 * hi) * 32 + ncol] = (_Float16)s[r];
}

// Read the LDS tile back in the 16-bit WMMA A layout: lane m = lane&15,
// halves [0..7] = K = 8*hi+{0..7}, halves [8..15] = K = 16+8*hi+{0..7}.
// Two ds_load_b128 per lane, landed directly into the operand register pair.
__device__ inline v16h gather_A(const _Float16* sbuf) {
  const int lane = threadIdx.x & 31;
  const int m    = lane & 15;
  const int k0   = (lane >> 4) * 8;
  union { v16h v; v8h h[2]; } a;
  a.h[0] = *(const v8h*)(sbuf + m * 32 + k0);
  a.h[1] = *(const v8h*)(sbuf + m * 32 + 16 + k0);
  return a.v;
}

__global__ __launch_bounds__(32) void snn_scan_kernel(
    const float* __restrict__ x,   // [B, T, I]
    const float* __restrict__ W1,  // [H, I]
    const float* __restrict__ W2,  // [H, H]
    const float* __restrict__ W3,  // [H, H]
    const float* __restrict__ W4,  // [O, H]
    float* __restrict__ out) {     // [T,B,O] then [T,B,H], concatenated
  __shared__ __align__(16) _Float16 sbuf[16 * 32];  // 1KB transpose tile

  const int lane = threadIdx.x & 31;
  const int m    = lane & 15;       // batch row within tile / N column
  const int hi   = lane >> 4;
  const int k0   = hi * 8;
  const int b0   = blockIdx.x * 16; // batch tile base
  const int n    = lane & 15;       // output column for C/D-layout stores

  // ---- Weights -> register-resident WMMA B operands (f16, zero-padded) ----
  const v16h B1a = make_B(W1, SNN_I, SNN_I, SNN_H, 0);    // N tile 0..15
  const v16h B1b = make_B(W1, SNN_I, SNN_I, SNN_H, 16);   // N tile 16..23 (+pad)
  const v16h B2a = make_B(W2, SNN_H, SNN_H, SNN_H, 0);
  const v16h B2b = make_B(W2, SNN_H, SNN_H, SNN_H, 16);
  const v16h B3a = make_B(W3, SNN_H, SNN_H, SNN_H, 0);
  const v16h B3b = make_B(W3, SNN_H, SNN_H, SNN_H, 16);
  const v16h B4  = make_B(W4, SNN_H, SNN_H, SNN_O, 0);    // only n<4 nonzero

  // ---- Membrane + spike state, register-resident in WMMA C/D layout ----
  v8f m1a = {}, m1b = {}, m2a = {}, m2b = {}, m3a = {}, m3b = {}, m4 = {};
  v8f s1a = {}, s1b = {}, s2a = {}, s2b = {}, s3a = {}, s3b = {};
  const v8f zero = {};

  // Per-lane x stream: lane loads x[b0+m][t][k0 .. k0+7] (32B) each step.
  const float* __restrict__ xrow = x + (size_t)(b0 + m) * SNN_T * SNN_I + k0;

  // Per-lane output pointers, advanced by one timestep each iteration;
  // row index r becomes an immediate offset (r*24 / r*4 floats) in the store.
  float* __restrict__ pm4 = out + ((size_t)(b0 + 8 * hi)) * SNN_O + n;
  float* __restrict__ ps3 = out + (size_t)SNN_T * SNN_B * SNN_O
                                + ((size_t)(b0 + 8 * hi)) * SNN_H + n;

  // Software pipeline: preload t=0.
  float4 cu0 = *(const float4*)(xrow + 0);
  float4 cu1 = *(const float4*)(xrow + 4);

  for (int t = 0; t < SNN_T; ++t) {
    // Issue next step's x load before this step's dependent chain.
    float4 nu0 = cu0, nu1 = cu1;
    if (t + 1 < SNN_T) {
      nu0 = *(const float4*)(xrow + (size_t)(t + 1) * SNN_I);
      nu1 = *(const float4*)(xrow + (size_t)(t + 1) * SNN_I + 4);
    }
    if (t + 32 < SNN_T)
      __builtin_prefetch(xrow + (size_t)(t + 32) * SNN_I, 0, 3);

    // ---- A1 from x (K=0..15 real, 16..31 zero-padded) ----
    v16h A1 = {};
    {
      const float xv[8] = {cu0.x, cu0.y, cu0.z, cu0.w, cu1.x, cu1.y, cu1.z, cu1.w};
#pragma unroll
      for (int j = 0; j < 8; ++j) A1[j] = (_Float16)xv[j];
    }

    // ---- Layer 1 ----
    v8f c1a = __builtin_amdgcn_wmma_f32_16x16x32_f16(false, A1, false, B1a,
                                                     (short)0, zero, false, false);
    v8f c1b = __builtin_amdgcn_wmma_f32_16x16x32_f16(false, A1, false, B1b,
                                                     (short)0, zero, false, false);
    lif_step(m1a, s1a, c1a, 0.9f);
    lif_step(m1b, s1b, c1b, 0.9f);

    // ---- transpose s1 -> A2 through LDS ----
    spill_spikes(sbuf, s1a, 0);
    spill_spikes(sbuf, s1b, 16);
    __syncthreads();
    const v16h A2 = gather_A(sbuf);
    __syncthreads();

    // ---- Layer 2 ----
    v8f c2a = __builtin_amdgcn_wmma_f32_16x16x32_f16(false, A2, false, B2a,
                                                     (short)0, zero, false, false);
    v8f c2b = __builtin_amdgcn_wmma_f32_16x16x32_f16(false, A2, false, B2b,
                                                     (short)0, zero, false, false);
    lif_step(m2a, s2a, c2a, 0.9f);
    lif_step(m2b, s2b, c2b, 0.9f);

    // ---- transpose s2 -> A3 ----
    spill_spikes(sbuf, s2a, 0);
    spill_spikes(sbuf, s2b, 16);
    __syncthreads();
    const v16h A3 = gather_A(sbuf);
    __syncthreads();

    // ---- Layer 3 ----
    v8f c3a = __builtin_amdgcn_wmma_f32_16x16x32_f16(false, A3, false, B3a,
                                                     (short)0, zero, false, false);
    v8f c3b = __builtin_amdgcn_wmma_f32_16x16x32_f16(false, A3, false, B3b,
                                                     (short)0, zero, false, false);
    lif_step(m3a, s3a, c3a, 0.9f);
    lif_step(m3b, s3b, c3b, 0.9f);

    // ---- transpose s3 -> A4 ----
    spill_spikes(sbuf, s3a, 0);
    spill_spikes(sbuf, s3b, 16);
    __syncthreads();
    const v16h A4 = gather_A(sbuf);
    __syncthreads();

    // ---- Layer 4: thresh=1e7 -> never resets; m4 = 0.95*m4 + cur4 ----
    v8f c4 = __builtin_amdgcn_wmma_f32_16x16x32_f16(false, A4, false, B4,
                                                    (short)0, zero, false, false);
#pragma unroll
    for (int e = 0; e < 8; ++e) m4[e] = __builtin_fmaf(0.95f, m4[e], c4[e]);

    // ---- record outputs: s3 [T,B,24] and m4 [T,B,4] (post-update) ----
#pragma unroll
    for (int r = 0; r < 8; ++r)
      ps3[r * SNN_H] = s3a[r];                 // columns n = 0..15
    if (n < 8) {
#pragma unroll
      for (int r = 0; r < 8; ++r)
        ps3[r * SNN_H + 16] = s3b[r];          // columns n = 16..23
    }
    if (n < SNN_O) {
#pragma unroll
      for (int r = 0; r < 8; ++r)
        pm4[r * SNN_O] = m4[r];
    }
    ps3 += (size_t)SNN_B * SNN_H;
    pm4 += (size_t)SNN_B * SNN_O;

    cu0 = nu0;
    cu1 = nu1;
  }
}

extern "C" void kernel_launch(void* const* d_in, const int* in_sizes, int n_in,
                              void* d_out, int out_size, void* d_ws, size_t ws_size,
                              hipStream_t stream) {
  (void)in_sizes; (void)n_in; (void)out_size; (void)d_ws; (void)ws_size;
  const float* x  = (const float*)d_in[0];
  const float* W1 = (const float*)d_in[1];
  const float* W2 = (const float*)d_in[2];
  const float* W3 = (const float*)d_in[3];
  const float* W4 = (const float*)d_in[4];
  float* out = (float*)d_out;

  dim3 grid(SNN_B / 16);  // 32 waves, each owns a 16-batch tile for all T steps
  dim3 block(32);         // one wave32 per workgroup (LDS tile wave-private)
  snn_scan_kernel<<<grid, block, 0, stream>>>(x, W1, W2, W3, W4, out);
}
